// SelfAttention_22119081574638
// MI455X (gfx1250) — matrix operand order repaired and checked by reading, not compiled, not run on hardware
//
#include <hip/hip_runtime.h>
#include <hip/hip_bf16.h>

#define USE_TDM 1   // flip to 0 to fall back to plain cooperative copies

// ---------------------------------------------------------------------------
// Problem constants
// ---------------------------------------------------------------------------
#define HIDDEN   1024
#define N_HEADS  16
#define HEAD_DIM 64
#define N_BLOCKS 16
#define BATCH    2
#define SEQ      4096
#define BLK      256
#define MTOT     (BATCH * SEQ)          // 8192

typedef __attribute__((ext_vector_type(4)))  __bf16 v4bf;
typedef __attribute__((ext_vector_type(8)))  __bf16 v8bf;
typedef __attribute__((ext_vector_type(16))) __bf16 v16bf;
typedef __attribute__((ext_vector_type(8)))  float  v8f;
typedef __attribute__((ext_vector_type(4)))  unsigned int u32x4;
typedef __attribute__((ext_vector_type(8)))  int i32x8;
typedef __attribute__((ext_vector_type(4)))  int i32x4;

static __device__ inline v16bf cat8(v8bf a, v8bf b) {
    v16bf r;
#pragma unroll
    for (int i = 0; i < 8; ++i) { r[i] = a[i]; r[i + 8] = b[i]; }
    return r;
}
static __device__ inline v8f zero8() {
    v8f z = {0.f, 0.f, 0.f, 0.f, 0.f, 0.f, 0.f, 0.f};
    return z;
}
static __device__ inline v8f wmma_bf16(v16bf a, v16bf b, v8f c) {
    return __builtin_amdgcn_wmma_f32_16x16x32_bf16(false, a, false, b,
                                                   (short)0, c, false, false);
}
// A-fragment (16x32 bf16): per-lane 16 contiguous K values.
static __device__ inline v16bf ldfrag_a(const __bf16* p) {
    return cat8(*(const v8bf*)p, *(const v8bf*)(p + 8));
}
// B-fragment (32x16 bf16): per-lane 16 contiguous K values.
static __device__ inline v16bf ldfrag_b(const __bf16* p) {
    return cat8(*(const v8bf*)p, *(const v8bf*)(p + 8));
}

// ---------------------------------------------------------------------------
// TDM: async 2D bf16 tile load global -> LDS with hardware row padding.
// Descriptor per CDNA5 ISA.  tensor dims == tile dims
// (no OOB), data_size=2B, pad_enable=1.
// ---------------------------------------------------------------------------
#if USE_TDM
static __device__ inline void tdm_load_2d_bf16(unsigned lds_addr, const void* gptr,
                                               unsigned tile_d0, unsigned tile_d1,
                                               unsigned stride_elems,
                                               unsigned pad_interval, unsigned pad_amount)
{
    unsigned long long ga = (unsigned long long)gptr;
    u32x4 g0;
    g0[0] = 1u;                                    // count=1, user mode
    g0[1] = lds_addr;                              // LDS byte address
    g0[2] = (unsigned)ga;                          // global_addr[31:0]
    g0[3] = (unsigned)((ga >> 32) & 0x01ffffffu)   // global_addr[56:32]
            | (2u << 30);                          // type = 2 ("image")
    i32x8 g1;
    g1[0] = (int)((1u << 16)                       // data_size = 1 -> 2 bytes
                | (1u << 20)                       // pad_enable
                | (pad_interval << 22)
                | (pad_amount   << 25));
    g1[1] = (int)((tile_d0 & 0xffffu) << 16);      // tensor_dim0[15:0]
    g1[2] = (int)(((tile_d0 >> 16) & 0xffffu)
                | ((tile_d1 & 0xffffu) << 16));    // tensor_dim0 hi | tensor_dim1 lo
    g1[3] = (int)(((tile_d1 >> 16) & 0xffffu)
                | ((tile_d0 & 0xffffu) << 16));    // tensor_dim1 hi | tile_dim0
    g1[4] = (int)(tile_d1 & 0xffffu);              // tile_dim1 (tile_dim2=0)
    g1[5] = (int)stride_elems;                     // tensor_dim0_stride[31:0]
    g1[6] = 0;
    g1[7] = 0;
    i32x4 gz; gz[0] = 0; gz[1] = 0; gz[2] = 0; gz[3] = 0;
#if defined(__clang_major__) && (__clang_major__ >= 23)
    i32x8 gz8;
#pragma unroll
    for (int i = 0; i < 8; ++i) gz8[i] = 0;
    __builtin_amdgcn_tensor_load_to_lds(g0, g1, gz, gz, gz8, 0);
#else
    __builtin_amdgcn_tensor_load_to_lds(g0, g1, gz, gz, 0);
#endif
}
#endif

static __device__ inline unsigned lds_off(const void* p) {
    return (unsigned)(unsigned long long)p;   // addr[31:0] = LDS offset (ISA §10.2)
}

// ---------------------------------------------------------------------------
// Kernel 0a: x f32 -> bf16   (8192x1024)
// ---------------------------------------------------------------------------
__global__ __launch_bounds__(256) void xcvt_kernel(const float* __restrict__ x,
                                                   __bf16* __restrict__ xb)
{
    size_t i = ((size_t)blockIdx.x * 256 + threadIdx.x) * 4;
    float4 v = *(const float4*)(x + i);
    v4bf o; o[0] = (__bf16)v.x; o[1] = (__bf16)v.y; o[2] = (__bf16)v.z; o[3] = (__bf16)v.w;
    *(v4bf*)(xb + i) = o;
}

// ---------------------------------------------------------------------------
// Kernel 0b: wT[z][n][k] = (bf16) w[z][k][n]
// ---------------------------------------------------------------------------
__global__ __launch_bounds__(256) void wtrans_kernel(
    const float* __restrict__ wq, const float* __restrict__ wk,
    const float* __restrict__ wv, __bf16* __restrict__ wtb)
{
    const float* w = (blockIdx.z == 0) ? wq : (blockIdx.z == 1) ? wk : wv;
    __bf16* out = wtb + (size_t)blockIdx.z * HIDDEN * HIDDEN;

    __shared__ __align__(16) __bf16 T[32 * 36];

    const int tid = threadIdx.x;
    const int k0 = blockIdx.y * 32;
    const int n0 = blockIdx.x * 32;

    {
        int r  = tid >> 3;
        int c4 = (tid & 7) * 4;
        float4 v = *(const float4*)(w + (size_t)(k0 + r) * HIDDEN + n0 + c4);
        T[(c4 + 0) * 36 + r] = (__bf16)v.x;
        T[(c4 + 1) * 36 + r] = (__bf16)v.y;
        T[(c4 + 2) * 36 + r] = (__bf16)v.z;
        T[(c4 + 3) * 36 + r] = (__bf16)v.w;
    }
    __syncthreads();
    {
        int row = tid >> 3;
        int c4  = (tid & 7) * 4;
        v4bf o;
#pragma unroll
        for (int j = 0; j < 4; ++j) o[j] = T[row * 36 + c4 + j];
        *(v4bf*)(out + (size_t)(n0 + row) * HIDDEN + k0 + c4) = o;
    }
}

// ---------------------------------------------------------------------------
// Kernel 1: QKV GEMM, bf16 in / bf16 out (f32 accum), TDM double-buffered.
// ---------------------------------------------------------------------------
#define TM 128
#define TN 128
#define TK 32
#define AP 40
#define TILE_E (TM * AP)

__global__ __launch_bounds__(256) void qkv_gemm_kernel(
    const __bf16* __restrict__ xb, const __bf16* __restrict__ wtb,
    const float* __restrict__ bq, const float* __restrict__ bk, const float* __restrict__ bv,
    __bf16* __restrict__ oq, __bf16* __restrict__ ok, __bf16* __restrict__ ov)
{
    const float* bias; __bf16* out;
    if      (blockIdx.z == 0) { bias = bq; out = oq; }
    else if (blockIdx.z == 1) { bias = bk; out = ok; }
    else                      { bias = bv; out = ov; }
    const __bf16* wt = wtb + (size_t)blockIdx.z * HIDDEN * HIDDEN;

    __shared__ __align__(16) __bf16 As[2 * TILE_E];
    __shared__ __align__(16) __bf16 Bs[2 * TILE_E];

    const int tid  = threadIdx.x;
    const int lane = tid & 31;
    const int wid  = tid >> 5;
    const int wy   = wid >> 1;
    const int wx   = wid & 1;
    const int lh   = lane >> 4;
    const int ll   = lane & 15;
    const int m0 = blockIdx.y * TM;
    const int n0 = blockIdx.x * TN;

    const __bf16* agp = xb + (size_t)m0 * HIDDEN;
    const __bf16* bgp = wt + (size_t)n0 * HIDDEN;

#if USE_TDM
    const unsigned aL = lds_off(As), bL = lds_off(Bs);
    if (wid == 0) {
        tdm_load_2d_bf16(aL, agp, TK, TM, HIDDEN, 3, 3);
        tdm_load_2d_bf16(bL, bgp, TK, TM, HIDDEN, 3, 3);
    }
    __builtin_amdgcn_s_wait_tensorcnt((short)0);
#else
#pragma unroll
    for (int i = 0; i < 2; ++i) {
        int c = tid + 256 * i, row = c >> 2, seg = c & 3;
        *(v8bf*)(As + row * AP + seg * 8) = *(const v8bf*)(agp + (size_t)row * HIDDEN + seg * 8);
        *(v8bf*)(Bs + row * AP + seg * 8) = *(const v8bf*)(bgp + (size_t)row * HIDDEN + seg * 8);
    }
#endif
    __syncthreads();

    float biasv[4];
#pragma unroll
    for (int u = 0; u < 4; ++u) biasv[u] = bias[n0 + wx * 64 + u * 16 + ll];

    v8f acc[2][4];
#pragma unroll
    for (int i = 0; i < 2; ++i)
#pragma unroll
        for (int u = 0; u < 4; ++u) acc[i][u] = zero8();

    for (int it = 0; it < HIDDEN / TK; ++it) {
        const int cur = it & 1;
        const int nxt = 1 - cur;
        if (it + 1 < HIDDEN / TK) {
            const int kn = (it + 1) * TK;
#if USE_TDM
            if (wid == 0) {
                tdm_load_2d_bf16(aL + nxt * TILE_E * 2, agp + kn, TK, TM, HIDDEN, 3, 3);
                tdm_load_2d_bf16(bL + nxt * TILE_E * 2, bgp + kn, TK, TM, HIDDEN, 3, 3);
            }
#else
#pragma unroll
            for (int i = 0; i < 2; ++i) {
                int c = tid + 256 * i, row = c >> 2, seg = c & 3;
                *(v8bf*)(As + nxt * TILE_E + row * AP + seg * 8) =
                    *(const v8bf*)(agp + (size_t)row * HIDDEN + kn + seg * 8);
                *(v8bf*)(Bs + nxt * TILE_E + row * AP + seg * 8) =
                    *(const v8bf*)(bgp + (size_t)row * HIDDEN + kn + seg * 8);
            }
#endif
        }
        const __bf16* asb = As + cur * TILE_E;
        const __bf16* bsb = Bs + cur * TILE_E;

        // hoist ALL fragments before the WMMA burst so the 8 matrix ops issue
        // back-to-back with partial dscnt waits instead of full round-trips
        v16bf af[2], bf[4];
#pragma unroll
        for (int i = 0; i < 2; ++i)
            af[i] = ldfrag_a(asb + (wy * 32 + i * 16 + ll) * AP + lh * 16);
#pragma unroll
        for (int u = 0; u < 4; ++u)
            bf[u] = ldfrag_b(bsb + (wx * 64 + u * 16 + ll) * AP + lh * 16);
#pragma unroll
        for (int u = 0; u < 4; ++u)
#pragma unroll
            for (int i = 0; i < 2; ++i)
                acc[i][u] = wmma_bf16(af[i], bf[u], acc[i][u]);

#if USE_TDM
        __builtin_amdgcn_s_wait_tensorcnt((short)0);   // no-op for non-issuing waves
#endif
        __syncthreads();
    }

#pragma unroll
    for (int i = 0; i < 2; ++i)
#pragma unroll
        for (int u = 0; u < 4; ++u)
#pragma unroll
            for (int r = 0; r < 8; ++r) {
                int m = m0 + wy * 32 + i * 16 + lh * 8 + r;
                int n = n0 + wx * 64 + u * 16 + ll;
                out[(size_t)m * HIDDEN + n] = (__bf16)(acc[i][u][r] + biasv[u]);
            }
}

// ---------------------------------------------------------------------------
// Kernel 2: block-local attention with permuted K/V.
// ---------------------------------------------------------------------------
#define KSP 72
#define VTP 264
#define KS_E (BLK * KSP)
#define VT_E (HEAD_DIM * VTP)
#define PW_E (8 * 16 * VTP)

__global__ __launch_bounds__(256) void block_attn_kernel(
    const __bf16* __restrict__ qb, const __bf16* __restrict__ kb,
    const __bf16* __restrict__ vb, const int* __restrict__ perm,
    float* __restrict__ out)
{
    extern __shared__ __align__(16) __bf16 smem[];
    __bf16* Ks = smem;                // [kv][d]  pitch 72
    __bf16* Vt = smem + KS_E;         // [d][kv]  pitch 264
    __bf16* Pw = smem + KS_E + VT_E;  // [wave][16][264]

    const int b    = blockIdx.z;
    const int flat = blockIdx.y;
    const int g = flat >> 4, n = flat & 15;
    const int qh   = blockIdx.x;

    const int tid  = threadIdx.x;
    const int lane = tid & 31;
    const int wid  = tid >> 5;
    const int lh   = lane >> 4;
    const int ll   = lane & 15;

    const int pidx = perm[flat];
    const int g2 = pidx >> 4, n2 = pidx & 15;
    const size_t kvbase = ((size_t)b * SEQ + (size_t)g2 * BLK) * HIDDEN + n2 * HEAD_DIM;

#if USE_TDM
    if (wid == 0)
        tdm_load_2d_bf16(lds_off(Ks), kb + kvbase, HEAD_DIM, BLK, HIDDEN, 4, 3);
#else
#pragma unroll
    for (int i = 0; i < 8; ++i) {
        int c = tid + 256 * i;
        int kv = c >> 3, seg = c & 7;
        *(v8bf*)(Ks + kv * KSP + seg * 8) =
            *(const v8bf*)(kb + kvbase + (size_t)kv * HIDDEN + seg * 8);
    }
#endif
    // V^T (element transpose by all threads; overlaps the K DMA)
#pragma unroll
    for (int i = 0; i < 16; ++i) {
        int s  = tid + 256 * i;
        int kv = s >> 4;
        int d4 = (s & 15) * 4;
        v4bf v = *(const v4bf*)(vb + kvbase + (size_t)kv * HIDDEN + d4);
#pragma unroll
        for (int j = 0; j < 4; ++j) Vt[(d4 + j) * VTP + kv] = v[j];
    }
#if USE_TDM
    __builtin_amdgcn_s_wait_tensorcnt((short)0);
#endif
    __syncthreads();

    const int    m0    = qh * 128 + wid * 16;
    const size_t qrow0 = (size_t)b * SEQ + (size_t)g * BLK + m0;

    v16bf aq[2];
#pragma unroll
    for (int c = 0; c < 2; ++c)
        aq[c] = ldfrag_a(qb + (qrow0 + ll) * HIDDEN + n * HEAD_DIM + c * 32 + lh * 16);

    // logits: software-pipelined by one tile (load t+1 frags during wmma on t)
    v8f sacc[16];
    v16bf bk0 = ldfrag_b(Ks + (0 * 16 + ll) * KSP + 0 * 32 + lh * 16);
    v16bf bk1 = ldfrag_b(Ks + (0 * 16 + ll) * KSP + 1 * 32 + lh * 16);
#pragma unroll
    for (int t = 0; t < 16; ++t) {
        v16bf c0 = bk0, c1 = bk1;
        if (t + 1 < 16) {
            bk0 = ldfrag_b(Ks + ((t + 1) * 16 + ll) * KSP + 0 * 32 + lh * 16);
            bk1 = ldfrag_b(Ks + ((t + 1) * 16 + ll) * KSP + 1 * 32 + lh * 16);
        }
        v8f a = zero8();
        a = wmma_bf16(aq[0], c0, a);
        a = wmma_bf16(aq[1], c1, a);
        sacc[t] = a * 0.125f;
    }

    float rmax[8], rsum[8];
#pragma unroll
    for (int r = 0; r < 8; ++r) {
        float m = -3.4e38f;
#pragma unroll
        for (int t = 0; t < 16; ++t) m = fmaxf(m, sacc[t][r]);
#pragma unroll
        for (int s = 8; s >= 1; s >>= 1) m = fmaxf(m, __shfl_xor(m, s, 32));
        rmax[r] = m;
        rsum[r] = 0.f;
    }

    __bf16* pw = Pw + wid * 16 * VTP;
#pragma unroll
    for (int t = 0; t < 16; ++t)
#pragma unroll
        for (int r = 0; r < 8; ++r) {
            float p = __expf(sacc[t][r] - rmax[r]);
            rsum[r] += p;
            pw[(r + lh * 8) * VTP + t * 16 + ll] = (__bf16)p;
        }
#pragma unroll
    for (int r = 0; r < 8; ++r) {
        float s = rsum[r];
#pragma unroll
        for (int x = 8; x >= 1; x >>= 1) s += __shfl_xor(s, x, 32);
        rsum[r] = s;
    }

    // ctx = P @ V : hoist the 4 V fragments per K-chunk before the WMMA burst
    v8f cacc[4];
#pragma unroll
    for (int u = 0; u < 4; ++u) cacc[u] = zero8();
#pragma unroll
    for (int kc = 0; kc < 8; ++kc) {
        v16bf af = ldfrag_a(pw + ll * VTP + kc * 32 + lh * 16);
        v16bf bv4[4];
#pragma unroll
        for (int u = 0; u < 4; ++u)
            bv4[u] = ldfrag_b(Vt + (u * 16 + ll) * VTP + kc * 32 + lh * 16);
#pragma unroll
        for (int u = 0; u < 4; ++u)
            cacc[u] = wmma_bf16(af, bv4[u], cacc[u]);
    }

#pragma unroll
    for (int r = 0; r < 8; ++r) {
        float inv = 1.0f / rsum[r];
        size_t row = (size_t)b * SEQ + (size_t)g * BLK + m0 + lh * 8 + r;
        float* o = out + row * HIDDEN + n * HEAD_DIM;
#pragma unroll
        for (int u = 0; u < 4; ++u)
            o[u * 16 + ll] = cacc[u][r] * inv;
    }
}

// ---------------------------------------------------------------------------
// Host launcher
// ---------------------------------------------------------------------------
extern "C" void kernel_launch(void* const* d_in, const int* in_sizes, int n_in,
                              void* d_out, int out_size, void* d_ws, size_t ws_size,
                              hipStream_t stream) {
    (void)in_sizes; (void)n_in; (void)out_size; (void)ws_size;

    const float* x    = (const float*)d_in[0];
    const int*   perm = (const int*)d_in[2];
    const float* qw   = (const float*)d_in[3];
    const float* qbi  = (const float*)d_in[4];
    const float* kw   = (const float*)d_in[5];
    const float* kbi  = (const float*)d_in[6];
    const float* vw   = (const float*)d_in[7];
    const float* vbi  = (const float*)d_in[8];

    const size_t NE = (size_t)MTOT * HIDDEN;
    __bf16* qbuf = (__bf16*)d_ws;
    __bf16* kbuf = qbuf + NE;
    __bf16* vbuf = kbuf + NE;
    __bf16* xb   = vbuf + NE;
    __bf16* wtb  = xb + NE;

    xcvt_kernel<<<dim3(MTOT * HIDDEN / 1024), 256, 0, stream>>>(x, xb);
    wtrans_kernel<<<dim3(32, 32, 3), 256, 0, stream>>>(qw, kw, vw, wtb);

    qkv_gemm_kernel<<<dim3(HIDDEN / TN, MTOT / TM, 3), 256, 0, stream>>>(
        xb, wtb, qbi, kbi, vbi, qbuf, kbuf, vbuf);

    const size_t shmem = (size_t)(KS_E + VT_E + PW_E) * sizeof(__bf16);   // 138240 B
    (void)hipFuncSetAttribute((const void*)block_attn_kernel,
                              hipFuncAttributeMaxDynamicSharedMemorySize, (int)shmem);
    block_attn_kernel<<<dim3(2, N_BLOCKS * N_HEADS, BATCH), 256, shmem, stream>>>(
        qbuf, kbuf, vbuf, perm, (float*)d_out);
}
